// QuantumRNN_83846351553151
// MI455X (gfx1250) — compile-verified, matrix operand order checked by reading, not run on hardware
//
#include <hip/hip_runtime.h>

// Problem constants
#define BB   32
#define TT   2048
#define FF   64
#define HH   256
#define NG   768      // 3*H
#define OUTN 64
#define HSTR 272      // padded LDS row stride (bf16 elems) to break bank conflicts

typedef __bf16 bf16;
typedef __attribute__((ext_vector_type(16))) __bf16 v16bf;
typedef __attribute__((ext_vector_type(8)))  __bf16 v8bf;
typedef __attribute__((ext_vector_type(8)))  float  v8f;

// Explicit global (addrspace 1) pointers so laundered weight loads lower as
// global_load_b128 (LOADcnt only), not flat_load_b128 (LOADcnt + DScnt).
typedef __attribute__((address_space(1))) const __bf16 gc_bf16;
typedef __attribute__((address_space(1))) const v16bf gc_v16bf;

// ---------------------------------------------------------------------------
// A-fragment loader for V_WMMA_F32_16X16X32_BF16.
// ISA layout (16-bit A 16x32): lane L (m = L%16, half = L/16):
//   elems 0..7  = K[half*8      .. half*8+7]
//   elems 8..15 = K[16+half*8   .. 16+half*8+7]
// => two aligned 16-byte loads from a row-major K-contiguous row.
// ---------------------------------------------------------------------------
static __device__ __forceinline__ v16bf load_afrag(const bf16* row_kt, int half) {
  v8bf lo = *(const v8bf*)(row_kt + half * 8);
  v8bf hi = *(const v8bf*)(row_kt + 16 + half * 8);
  return __builtin_shufflevector(lo, hi, 0,1,2,3,4,5,6,7,8,9,10,11,12,13,14,15);
}

// B-fragment loader: 32 contiguous bytes at a compile-time element offset from
// a per-wave base pointer. The base is laundered once per timestep in the main
// loop, so these stay in-loop global_load_b128 with immediate offsets.
static __device__ __forceinline__ v16bf load_bfrag(gc_bf16* base, int elem_off) {
  return *(gc_v16bf*)(base + elem_off);
}

static __device__ __forceinline__ v8f wmma_bf16(v16bf a, v16bf b, v8f c) {
  return __builtin_amdgcn_wmma_f32_16x16x32_bf16(false, a, false, b, (short)0, c,
                                                 false, false);
}

static __device__ __forceinline__ float sigmoidf_(float x) {
  return 1.0f / (1.0f + __expf(-x));
}

// ---------------------------------------------------------------------------
// LayerNorm over F=64 features -> bf16. One wave per row, 2 elems per lane.
// ---------------------------------------------------------------------------
__global__ __launch_bounds__(256) void ln_kernel(const float* __restrict__ x,
                                                 const float* __restrict__ gamma,
                                                 const float* __restrict__ beta,
                                                 bf16* __restrict__ xn) {
  const int row  = blockIdx.x * 8 + (threadIdx.x >> 5);
  const int lane = threadIdx.x & 31;
  const float2 v = *(const float2*)(x + (size_t)row * FF + lane * 2);
  float s  = v.x + v.y;
  float ss = v.x * v.x + v.y * v.y;
  #pragma unroll
  for (int o = 16; o > 0; o >>= 1) {
    s  += __shfl_xor(s,  o, 32);
    ss += __shfl_xor(ss, o, 32);
  }
  const float mu  = s * (1.0f / 64.0f);
  const float var = ss * (1.0f / 64.0f) - mu * mu;
  const float inv = rsqrtf(var + 1e-3f);
  const int f = lane * 2;
  const float y0 = (v.x - mu) * inv * gamma[f]     + beta[f];
  const float y1 = (v.y - mu) * inv * gamma[f + 1] + beta[f + 1];
  bf16* o = xn + (size_t)row * FF + f;
  o[0] = (bf16)y0;
  o[1] = (bf16)y1;
}

// ---------------------------------------------------------------------------
// Pack W[K x 768] (f32, row-major) into WMMA B-fragment order (bf16), keyed
// by (jtile, ktile, gate) so the 3 gate fragments of a (jt,kt) pair are
// contiguous and a wave addresses all of them with immediate offsets from a
// single base pointer:
//   packed[(((jt*KT + kt)*3 + g)*32 + lane)*16 + e]
//     = W[k(e,lane,kt)][jt*16 + g*256 + lane%16]
// ---------------------------------------------------------------------------
__global__ __launch_bounds__(256) void pack_kernel(const float* __restrict__ W,
                                                   bf16* __restrict__ out,
                                                   int KT, int total) {
  const int idx = blockIdx.x * 256 + threadIdx.x;
  if (idx >= total) return;
  const int e     = idx & 15;
  const int lane  = (idx >> 4) & 31;
  const int rest  = idx >> 9;
  const int g     = rest % 3;
  const int rest2 = rest / 3;
  const int kt    = rest2 % KT;
  const int jt    = rest2 / KT;
  const int half  = lane >> 4;
  const int v     = e >> 1;
  const int j     = e & 1;
  const int kb    = (v < 4) ? (2 * v + 8 * half) : (16 + 2 * (v - 4) + 8 * half);
  const int k     = kt * 32 + kb + j;
  const int n     = jt * 16 + g * 256 + (lane & 15);
  out[idx] = (bf16)W[(size_t)k * NG + n];
}

// ---------------------------------------------------------------------------
// Persistent fused 2-layer GRU. Single workgroup (1024 threads = 32 waves).
// Wave (mtile, jtile) owns output rows [mtile*16,+16) and gate columns
// {n0, 256+n0, 512+n0} with n0 = jtile*16 + lane%16 -> all gate math in regs.
// Hidden states live in LDS as bf16; weights stream from L2 every step.
// ---------------------------------------------------------------------------
__global__ __launch_bounds__(1024, 1) void gru_fused(
    const bf16* __restrict__ xn,
    const bf16* __restrict__ pk1, const bf16* __restrict__ prk1,
    const float* __restrict__ b1,
    const bf16* __restrict__ pk2, const bf16* __restrict__ prk2,
    const float* __restrict__ b2,
    float* __restrict__ h2out) {
  __shared__ bf16 h1[BB * HSTR];
  __shared__ bf16 h2[BB * HSTR];

  const int tid = threadIdx.x;
  for (int i = tid; i < BB * HSTR; i += 1024) {
    h1[i] = (bf16)0.0f;
    h2[i] = (bf16)0.0f;
  }
  __syncthreads();

  const int lane  = tid & 31;
  const int wave  = tid >> 5;
  const int mtile = wave & 1;    // 0..1  (rows 0-15 / 16-31)
  const int jtile = wave >> 1;   // 0..15 (H columns)
  const int half  = lane >> 4;
  const int nloc  = lane & 15;
  const int n0    = jtile * 16 + nloc;     // column within H
  const int ma    = mtile * 16 + nloc;     // A-fragment batch row

  // Bias registers (Keras reset_after: bias[0] on x-proj, bias[1] on inner)
  const float b1xz = b1[n0],          b1xr = b1[HH + n0],          b1xh = b1[2 * HH + n0];
  const float b1rz = b1[NG + n0],     b1rr = b1[NG + HH + n0],     b1rh = b1[NG + 2 * HH + n0];
  const float b2xz = b2[n0],          b2xr = b2[HH + n0],          b2xh = b2[2 * HH + n0];
  const float b2rz = b2[NG + n0],     b2rr = b2[NG + HH + n0],     b2rh = b2[NG + 2 * HH + n0];

  // One per-wave base pointer per packed array; fragments addressed by
  // compile-time immediate offsets (kt*3 + g)*512 elems (max 23KB).
  gc_bf16* w1 = (gc_bf16*)(size_t)(pk1  + (size_t)jtile * 2 * 3 * 512 + lane * 16);
  gc_bf16* u1 = (gc_bf16*)(size_t)(prk1 + (size_t)jtile * 8 * 3 * 512 + lane * 16);
  gc_bf16* w2 = (gc_bf16*)(size_t)(pk2  + (size_t)jtile * 8 * 3 * 512 + lane * 16);
  gc_bf16* u2 = (gc_bf16*)(size_t)(prk2 + (size_t)jtile * 8 * 3 * 512 + lane * 16);

  const bf16* arow_base = xn + (size_t)ma * TT * FF;

  for (int t = 0; t < TT; ++t) {
    // Launder the weight bases once per step: blocks LICM from hoisting the
    // (loop-invariant) fragment loads into a giant spill, while keeping them
    // schedulable global_load_b128 within the iteration.
    asm volatile("" : "+v"(w1), "+v"(u1), "+v"(w2), "+v"(u2));

    // ================= Layer 1 =================
    v8f xz = {}, xr = {}, xh = {}, rz = {}, rr = {}, rh = {};

    const bf16* arow = arow_base + (size_t)t * FF;
    // Prefetch activations 8 steps ahead (128B/step/row). global_prefetch_b8.
    if (t + 8 < TT) __builtin_prefetch(arow + 8 * FF, 0, 3);
    #pragma unroll
    for (int kt = 0; kt < 2; ++kt) {               // x-projection, K = 64
      const v16bf a = load_afrag(arow + kt * 32, half);
      xz = wmma_bf16(a, load_bfrag(w1, (kt * 3 + 0) * 512), xz);
      xr = wmma_bf16(a, load_bfrag(w1, (kt * 3 + 1) * 512), xr);
      xh = wmma_bf16(a, load_bfrag(w1, (kt * 3 + 2) * 512), xh);
    }
    #pragma unroll
    for (int kt = 0; kt < 8; ++kt) {               // recurrent, K = 256
      const v16bf a = load_afrag(&h1[ma * HSTR + kt * 32], half);
      rz = wmma_bf16(a, load_bfrag(u1, (kt * 3 + 0) * 512), rz);
      rr = wmma_bf16(a, load_bfrag(u1, (kt * 3 + 1) * 512), rr);
      rh = wmma_bf16(a, load_bfrag(u1, (kt * 3 + 2) * 512), rh);
    }
    float hn1[8];
    #pragma unroll
    for (int r = 0; r < 8; ++r) {
      const int row = mtile * 16 + r + 8 * half;   // D-tile: m = r + 8*half
      const float z  = sigmoidf_((xz[r] + b1xz) + (rz[r] + b1rz));
      const float rg = sigmoidf_((xr[r] + b1xr) + (rr[r] + b1rr));
      const float hh = tanhf((xh[r] + b1xh) + rg * (rh[r] + b1rh));
      const float hp = (float)h1[row * HSTR + n0];
      hn1[r] = z * hp + (1.0f - z) * hh;
    }
    __syncthreads();                               // all reads of h1_old done
    #pragma unroll
    for (int r = 0; r < 8; ++r) {
      const int row = mtile * 16 + r + 8 * half;
      h1[row * HSTR + n0] = (bf16)hn1[r];
    }
    __syncthreads();                               // h1_new visible

    // ================= Layer 2 =================
    v8f xz2 = {}, xr2 = {}, xh2 = {}, rz2 = {}, rr2 = {}, rh2 = {};
    #pragma unroll
    for (int kt = 0; kt < 8; ++kt) {               // x-projection from h1_new
      const v16bf a = load_afrag(&h1[ma * HSTR + kt * 32], half);
      xz2 = wmma_bf16(a, load_bfrag(w2, (kt * 3 + 0) * 512), xz2);
      xr2 = wmma_bf16(a, load_bfrag(w2, (kt * 3 + 1) * 512), xr2);
      xh2 = wmma_bf16(a, load_bfrag(w2, (kt * 3 + 2) * 512), xh2);
    }
    #pragma unroll
    for (int kt = 0; kt < 8; ++kt) {               // recurrent from h2_old
      const v16bf a = load_afrag(&h2[ma * HSTR + kt * 32], half);
      rz2 = wmma_bf16(a, load_bfrag(u2, (kt * 3 + 0) * 512), rz2);
      rr2 = wmma_bf16(a, load_bfrag(u2, (kt * 3 + 1) * 512), rr2);
      rh2 = wmma_bf16(a, load_bfrag(u2, (kt * 3 + 2) * 512), rh2);
    }
    float hn2[8];
    #pragma unroll
    for (int r = 0; r < 8; ++r) {
      const int row = mtile * 16 + r + 8 * half;
      const float z  = sigmoidf_((xz2[r] + b2xz) + (rz2[r] + b2rz));
      const float rg = sigmoidf_((xr2[r] + b2xr) + (rr2[r] + b2rr));
      const float hh = tanhf((xh2[r] + b2xh) + rg * (rh2[r] + b2rh));
      const float hp = (float)h2[row * HSTR + n0];
      hn2[r] = z * hp + (1.0f - z) * hh;
      if (t == TT - 1) h2out[row * HH + n0] = hn2[r];   // f32 final state
    }
    __syncthreads();                               // all reads of h2_old done
    #pragma unroll
    for (int r = 0; r < 8; ++r) {
      const int row = mtile * 16 + r + 8 * half;
      h2[row * HSTR + n0] = (bf16)hn2[r];
    }
    // next iteration's L1 barrier pair orders these writes vs. L2 reads
  }
}

// ---------------------------------------------------------------------------
// Final dense: [32,256] @ [256,64] + bias. Tiny -> plain VALU kernel.
// ---------------------------------------------------------------------------
__global__ __launch_bounds__(256) void dense_kernel(const float* __restrict__ h2f,
                                                    const float* __restrict__ wd,
                                                    const float* __restrict__ bd,
                                                    float* __restrict__ out) {
  const int idx = blockIdx.x * 256 + threadIdx.x;
  if (idx >= BB * OUTN) return;
  const int b = idx / OUTN;
  const int o = idx % OUTN;
  float s = bd[o];
  #pragma unroll 4
  for (int j = 0; j < HH; ++j) s += h2f[b * HH + j] * wd[j * OUTN + o];
  out[idx] = s;
}

// ---------------------------------------------------------------------------
extern "C" void kernel_launch(void* const* d_in, const int* in_sizes, int n_in,
                              void* d_out, int out_size, void* d_ws, size_t ws_size,
                              hipStream_t stream) {
  const float* x     = (const float*)d_in[0];
  const float* gamma = (const float*)d_in[1];
  const float* beta  = (const float*)d_in[2];
  const float* k1    = (const float*)d_in[3];
  const float* rk1   = (const float*)d_in[4];
  const float* b1    = (const float*)d_in[5];
  const float* k2    = (const float*)d_in[6];
  const float* rk2   = (const float*)d_in[7];
  const float* b2    = (const float*)d_in[8];
  const float* wd    = (const float*)d_in[9];
  const float* bd    = (const float*)d_in[10];
  float* out = (float*)d_out;

  // Workspace carve-up (~9.3 MB total, 256B-aligned chunks)
  char* w = (char*)d_ws;
  auto alloc = [&](size_t bytes) -> char* {
    char* p = w;
    w += (bytes + 255) & ~(size_t)255;
    return p;
  };
  bf16* xn   = (bf16*)alloc((size_t)BB * TT * FF * sizeof(bf16));   // 8 MB
  bf16* pk1  = (bf16*)alloc((size_t)FF * NG * sizeof(bf16));        // 96 KB
  bf16* prk1 = (bf16*)alloc((size_t)HH * NG * sizeof(bf16));        // 384 KB
  bf16* pk2  = (bf16*)alloc((size_t)HH * NG * sizeof(bf16));
  bf16* prk2 = (bf16*)alloc((size_t)HH * NG * sizeof(bf16));
  float* h2f = (float*)alloc((size_t)BB * HH * sizeof(float));      // 32 KB
  (void)ws_size; (void)in_sizes; (void)n_in; (void)out_size;

  // 1) LayerNorm -> bf16 activations
  ln_kernel<<<(BB * TT) / 8, 256, 0, stream>>>(x, gamma, beta, xn);

  // 2) Pack weights into WMMA B-fragment order (one-time cost per call)
  const int tot_k1 = 16 * 2 * 3 * 512;   // JT * KT * 3 * 512
  const int tot_hh = 16 * 8 * 3 * 512;
  pack_kernel<<<(tot_k1 + 255) / 256, 256, 0, stream>>>(k1,  pk1,  2, tot_k1);
  pack_kernel<<<(tot_hh + 255) / 256, 256, 0, stream>>>(rk1, prk1, 8, tot_hh);
  pack_kernel<<<(tot_hh + 255) / 256, 256, 0, stream>>>(k2,  pk2,  8, tot_hh);
  pack_kernel<<<(tot_hh + 255) / 256, 256, 0, stream>>>(rk2, prk2, 8, tot_hh);

  // 3) Persistent fused 2-layer GRU over T=2048 steps (single WGP, WMMA)
  gru_fused<<<1, 1024, 0, stream>>>(xn, pk1, prk1, b1, pk2, prk2, b2, h2f);

  // 4) Final dense head
  dense_kernel<<<(BB * OUTN + 255) / 256, 256, 0, stream>>>(h2f, wd, bd, out);
}
